// Attention_Rel_Scl_26276609917751
// MI455X (gfx1250) — matrix operand order repaired and checked by reading, not compiled, not run on hardware
//
#include <hip/hip_runtime.h>

// ---- problem constants ----
#define B_  4
#define S_  2048
#define D_  512
#define H_  8
#define DK_ 64

#define WROW 2056  // padded LDS row stride (floats): 2056*4 = 8224 B, 32B-aligned

typedef __attribute__((ext_vector_type(16))) _Float16 v16h;
typedef __attribute__((ext_vector_type(8)))  _Float16 v8h;
typedef __attribute__((ext_vector_type(8)))  float    v8f;

static __device__ __forceinline__ v16h combine8(v8h lo, v8h hi) {
  v16h r;
#pragma unroll
  for (int i = 0; i < 8; ++i) { r[i] = lo[i]; r[i + 8] = hi[i]; }
  return r;
}

static __device__ __forceinline__ v8f wmma_f16(v16h a, v16h b, v8f c) {
  return __builtin_amdgcn_wmma_f32_16x16x32_f16(false, a, false, b, (short)0, c,
                                                false, false);
}

// ---------------------------------------------------------------------------
// elementwise f32 -> f16 conversion
// ---------------------------------------------------------------------------
__global__ void f32_to_f16_kernel(const float* __restrict__ src,
                                  _Float16* __restrict__ dst, int n) {
  int i = blockIdx.x * blockDim.x + threadIdx.x;
  if (i < n) dst[i] = (_Float16)src[i];
}

// ---------------------------------------------------------------------------
// V (B*S, 512) f16 head-major  ->  Vt[b,h,d,j] f16  (PV B-operand contiguous in j)
// ---------------------------------------------------------------------------
__global__ void transpose_v_kernel(const _Float16* __restrict__ V,
                                   _Float16* __restrict__ Vt) {
  int idx = blockIdx.x * blockDim.x + threadIdx.x;  // over B*H*DK*S
  int j    = idx & (S_ - 1);
  int rest = idx >> 11;
  int d    = rest & (DK_ - 1);
  int bh   = rest >> 6;
  int b = bh >> 3, h = bh & 7;
  Vt[idx] = V[(size_t)(b * S_ + j) * D_ + h * DK_ + d];
}

// ---------------------------------------------------------------------------
// C[M,N] = A[M,K] @ W[K,N] + bias  (A,W f16 row-major, bias f32, OutT f16/f32)
// block: 256 threads = 8 waves; tile 128(M) x 64(N); k-step 32.
// ---------------------------------------------------------------------------
template <typename OutT>
__global__ void gemm_f16_kernel(const _Float16* __restrict__ A,
                                const _Float16* __restrict__ W,
                                const float* __restrict__ bias,
                                OutT* __restrict__ C, int M, int N, int K) {
  __shared__ _Float16 As[128][32];  // A tile (row-major)
  __shared__ _Float16 Bt[64][32];   // W tile transposed: Bt[n][k]

  const int tid  = threadIdx.x;
  const int lane = tid & 31;
  const int wave = tid >> 5;
  const int half = lane >> 4;
  const int l16  = lane & 15;
  const int n0 = blockIdx.x * 64;
  const int m0 = blockIdx.y * 128;

  v8f acc[4];
#pragma unroll
  for (int c = 0; c < 4; ++c)
#pragma unroll
    for (int i = 0; i < 8; ++i) acc[c][i] = 0.0f;

  for (int k0 = 0; k0 < K; k0 += 32) {
    {
      int r = tid >> 2;            // 0..63
      int q = (tid & 3) * 8;       // 0,8,16,24
      *(v8h*)&As[r][q]      = *(const v8h*)&A[(size_t)(m0 + r) * K + k0 + q];
      *(v8h*)&As[r + 64][q] = *(const v8h*)&A[(size_t)(m0 + r + 64) * K + k0 + q];
      if (k0 + 32 < K)
        __builtin_prefetch(&A[(size_t)(m0 + r) * K + k0 + 32 + q], 0, 3);
    }
    {
      int r  = tid >> 3;           // 0..31 (k)
      int c8 = (tid & 7) * 8;      // 0..56 (n)
      v8h wv = *(const v8h*)&W[(size_t)(k0 + r) * N + n0 + c8];
#pragma unroll
      for (int i = 0; i < 8; ++i) Bt[c8 + i][r] = wv[i];
    }
    __syncthreads();

    const int am = wave * 16 + l16;
    v16h a = combine8(*(const v8h*)&As[am][half * 8],
                      *(const v8h*)&As[am][16 + half * 8]);
#pragma unroll
    for (int c = 0; c < 4; ++c) {
      const int bn = c * 16 + l16;
      v16h b = combine8(*(const v8h*)&Bt[bn][half * 8],
                        *(const v8h*)&Bt[bn][16 + half * 8]);
      acc[c] = wmma_f16(a, b, acc[c]);
    }
    __syncthreads();
  }

  const int mbase = m0 + wave * 16 + half * 8;
#pragma unroll
  for (int c = 0; c < 4; ++c) {
    const int nn = n0 + c * 16 + l16;
    const float bb = bias[nn];
#pragma unroll
    for (int r = 0; r < 8; ++r)
      C[(size_t)(mbase + r) * N + nn] = (OutT)(acc[c][r] + bb);
  }
}

// ---------------------------------------------------------------------------
// Fused attention: scores -> softmax -> +rel_bias -> attn write -> PV -> ctx.
// Block = 128 threads (4 waves) handles (b, h, 16 query rows).
// Dynamic LDS: Wl[16][WROW] f32 scores block | inv[16] | Pl[4][16][64] partials.
// Each wave owns a 512-key column range for both score and PV phases.
// ---------------------------------------------------------------------------
__global__ void fused_attn_kernel(const _Float16* __restrict__ Q,
                                  const _Float16* __restrict__ Kh,
                                  const _Float16* __restrict__ Vt,
                                  const float* __restrict__ scale_p,
                                  const float* __restrict__ rel_table,
                                  float* __restrict__ attn,
                                  _Float16* __restrict__ ctx) {
  extern __shared__ float smem[];
  float* Wl   = smem;                       // 16 * WROW
  float* invb = smem + 16 * WROW;           // 16
  float* Pl   = invb + 16;                  // 4 * 16 * 64

  const int tid  = threadIdx.x;             // 0..127
  const int lane = tid & 31;
  const int wave = tid >> 5;                // 0..3
  const int half = lane >> 4;
  const int l16  = lane & 15;
  const int bh = blockIdx.y;
  const int b = bh >> 3, h = bh & 7;
  const int q0 = blockIdx.x * 16;
  const float scale = scale_p[0];
  const int jlo = wave * 512, jhi = jlo + 512;

  // ---- phase 1: scores = scale * Q K^T into LDS ----
  const _Float16* qrow = Q + ((size_t)(b * S_ + q0 + l16) * D_ + h * DK_);
  v16h a0 = combine8(*(const v8h*)&qrow[half * 8],      *(const v8h*)&qrow[16 + half * 8]);
  v16h a1 = combine8(*(const v8h*)&qrow[32 + half * 8], *(const v8h*)&qrow[48 + half * 8]);

  for (int j0 = jlo; j0 < jhi; j0 += 16) {
    const _Float16* krow = Kh + ((size_t)(b * S_ + j0 + l16) * D_ + h * DK_);
    v16h b0 = combine8(*(const v8h*)&krow[half * 8],      *(const v8h*)&krow[16 + half * 8]);
    v16h b1 = combine8(*(const v8h*)&krow[32 + half * 8], *(const v8h*)&krow[48 + half * 8]);
    v8f acc;
#pragma unroll
    for (int i = 0; i < 8; ++i) acc[i] = 0.0f;
    acc = wmma_f16(a0, b0, acc);
    acc = wmma_f16(a1, b1, acc);
#pragma unroll
    for (int r = 0; r < 8; ++r)
      Wl[(half * 8 + r) * WROW + j0 + l16] = acc[r] * scale;
  }
  __syncthreads();

  // ---- phase 2: softmax over LDS rows (8 lanes per row) ----
  {
    const int srow = tid >> 3;  // 0..15
    const int sub  = tid & 7;
    float* prow = Wl + srow * WROW;
    float m = -3.402823466e38f;
    for (int j = sub; j < S_; j += 8) m = fmaxf(m, prow[j]);
#pragma unroll
    for (int off = 4; off; off >>= 1) m = fmaxf(m, __shfl_xor(m, off, 32));
    float s = 0.0f;
    for (int j = sub; j < S_; j += 8) {
      float e = __expf(prow[j] - m);
      prow[j] = e;
      s += e;
    }
#pragma unroll
    for (int off = 4; off; off >>= 1) s += __shfl_xor(s, off, 32);
    if (sub == 0) invb[srow] = 1.0f / s;
  }
  __syncthreads();

  // ---- phase 3: normalize + rel bias; stream attn out (non-temporal: the
  //      global copy is never re-read, keep L2 for Q/K/V/ctx) ----
  for (int r = 0; r < 16; ++r) {
    const float iv = invb[r];
    const int ibase = (q0 + r + S_ - 1) * H_ + h;
    float* lrow = Wl + r * WROW;
    float* grow = attn + ((size_t)bh * S_ + q0 + r) * S_;
    for (int j = tid; j < S_; j += 128) {
      float w = lrow[j] * iv + rel_table[ibase - j * H_];
      lrow[j] = w;
      __builtin_nontemporal_store(w, &grow[j]);
    }
  }
  __syncthreads();

  // ---- phase 4: PV over this wave's key range, f32->f16 A frags from LDS ----
  v8f acc[4];
#pragma unroll
  for (int c = 0; c < 4; ++c)
#pragma unroll
    for (int i = 0; i < 8; ++i) acc[c][i] = 0.0f;

  const _Float16* vbase = Vt + (size_t)bh * DK_ * S_;
  for (int j0 = jlo; j0 < jhi; j0 += 32) {
    const float* ar = Wl + l16 * WROW + j0;
    v8f lo = *(const v8f*)&ar[half * 8];
    v8f hi = *(const v8f*)&ar[16 + half * 8];
    v16h a;
#pragma unroll
    for (int i = 0; i < 8; ++i) { a[i] = (_Float16)lo[i]; a[i + 8] = (_Float16)hi[i]; }
#pragma unroll
    for (int c = 0; c < 4; ++c) {
      const _Float16* vr = vbase + (size_t)(c * 16 + l16) * S_ + j0 + half * 8;
      v16h bf = combine8(*(const v8h*)&vr[0], *(const v8h*)&vr[16]);
      acc[c] = wmma_f16(a, bf, acc[c]);
    }
  }
  // stash per-wave partials
#pragma unroll
  for (int c = 0; c < 4; ++c)
#pragma unroll
    for (int r = 0; r < 8; ++r)
      Pl[wave * 1024 + (half * 8 + r) * 64 + c * 16 + l16] = acc[c][r];
  __syncthreads();

  // ---- reduce 4 partials, write f16 context ----
  {
    const int idx = tid * 8;          // 1024 elements / 128 threads
    const int m_ = idx >> 6;
    const int d_ = idx & 63;
    v8h o;
#pragma unroll
    for (int k = 0; k < 8; ++k) {
      float s = Pl[m_ * 64 + d_ + k] + Pl[1024 + m_ * 64 + d_ + k] +
                Pl[2048 + m_ * 64 + d_ + k] + Pl[3072 + m_ * 64 + d_ + k];
      o[k] = (_Float16)s;
    }
    *(v8h*)&ctx[(size_t)(b * S_ + q0 + m_) * D_ + h * DK_ + d_] = o;
  }
}

// ---------------------------------------------------------------------------
extern "C" void kernel_launch(void* const* d_in, const int* in_sizes, int n_in,
                              void* d_out, int out_size, void* d_ws, size_t ws_size,
                              hipStream_t stream) {
  (void)in_sizes; (void)n_in; (void)out_size; (void)ws_size;

  const float* x   = (const float*)d_in[0];
  const float* Wq  = (const float*)d_in[1];
  const float* bq  = (const float*)d_in[2];
  const float* Wk  = (const float*)d_in[3];
  const float* bk  = (const float*)d_in[4];
  const float* Wv  = (const float*)d_in[5];
  const float* bv  = (const float*)d_in[6];
  const float* Wo  = (const float*)d_in[7];
  const float* bo  = (const float*)d_in[8];
  const float* scale     = (const float*)d_in[9];
  const float* rel_table = (const float*)d_in[10];

  float* out  = (float*)d_out;                        // B*S*D
  float* attn = (float*)d_out + (size_t)B_ * S_ * D_; // B*H*S*S

  const size_t NX = (size_t)B_ * S_ * D_;  // 4194304
  const size_t NW = (size_t)D_ * D_;       // 262144

  _Float16* xh  = (_Float16*)d_ws;
  _Float16* Wqh = xh  + NX;
  _Float16* Wkh = Wqh + NW;
  _Float16* Wvh = Wkh + NW;
  _Float16* Woh = Wvh + NW;
  _Float16* Qh  = Woh + NW;
  _Float16* Kh  = Qh  + NX;
  _Float16* Vh  = Kh  + NX;
  _Float16* Vth = Vh  + NX;   // transposed V [b,h,d,j]
  _Float16* ctx = Vth + NX;

  // 1) conversions
  f32_to_f16_kernel<<<(int)(NX / 256), 256, 0, stream>>>(x,  xh,  (int)NX);
  f32_to_f16_kernel<<<(int)(NW / 256), 256, 0, stream>>>(Wq, Wqh, (int)NW);
  f32_to_f16_kernel<<<(int)(NW / 256), 256, 0, stream>>>(Wk, Wkh, (int)NW);
  f32_to_f16_kernel<<<(int)(NW / 256), 256, 0, stream>>>(Wv, Wvh, (int)NW);
  f32_to_f16_kernel<<<(int)(NW / 256), 256, 0, stream>>>(Wo, Woh, (int)NW);

  // 2) Q/K/V projections: (8192x512) @ (512x512) + bias -> f16
  dim3 ggrid(D_ / 64, (B_ * S_) / 128);
  gemm_f16_kernel<_Float16><<<ggrid, 256, 0, stream>>>(xh, Wqh, bq, Qh, B_ * S_, D_, D_);
  gemm_f16_kernel<_Float16><<<ggrid, 256, 0, stream>>>(xh, Wkh, bk, Kh, B_ * S_, D_, D_);
  gemm_f16_kernel<_Float16><<<ggrid, 256, 0, stream>>>(xh, Wvh, bv, Vh, B_ * S_, D_, D_);

  // 3) V transpose for PV B-operand
  transpose_v_kernel<<<(int)(NX / 256), 256, 0, stream>>>(Vh, Vth);

  // 4) fused scores/softmax/bias/PV; attn written exactly once (NT stores)
  const size_t lds_bytes = (size_t)(16 * WROW + 16 + 4 * 16 * 64) * sizeof(float);
  hipFuncSetAttribute(reinterpret_cast<const void*>(fused_attn_kernel),
                      hipFuncAttributeMaxDynamicSharedMemorySize, (int)lds_bytes);
  fused_attn_kernel<<<dim3(S_ / 16, B_ * H_), 128, lds_bytes, stream>>>(
      Qh, Kh, Vth, scale, rel_table, attn, ctx);

  // 5) out = ctx @ Wo + bo -> f32
  gemm_f16_kernel<float><<<ggrid, 256, 0, stream>>>(ctx, Woh, bo, out, B_ * S_, D_, D_);
}
